// GeATLayer_76622216561260
// MI455X (gfx1250) — matrix operand order repaired and checked by reading, not compiled, not run on hardware
//
#include <hip/hip_runtime.h>
#include <hip/hip_bf16.h>

typedef __attribute__((ext_vector_type(16))) _Float16 v16h;
typedef __attribute__((ext_vector_type(8)))  _Float16 v8h;
typedef __attribute__((ext_vector_type(8)))  float    v8f;
typedef __attribute__((ext_vector_type(4)))  int      v4i;

#define NDIM   64
#define NHEAD  4
#define NBOND  4
#define HIDD   64
#define NEG_SLOPE 0.2f

// Async global->LDS staging (gfx1250 GLOBAL_LOAD_ASYNC_TO_LDS_B128, ASYNCcnt).
#if __has_builtin(__builtin_amdgcn_global_load_async_to_lds_b128) && \
    __has_builtin(__builtin_amdgcn_s_wait_asynccnt)
#define USE_ASYNC_LDS 1
#else
#define USE_ASYNC_LDS 0
#endif

#if USE_ASYNC_LDS
typedef __attribute__((address_space(1))) v4i glob_v4i;
typedef __attribute__((address_space(3))) v4i lds_v4i;
static __device__ __forceinline__ void async_cp16(void* lds, const void* g) {
  __builtin_amdgcn_global_load_async_to_lds_b128(
      (glob_v4i*)g, (lds_v4i*)lds, 0, 0);
}
#endif

// ---------------------------------------------------------------- WMMA helpers
static __device__ __forceinline__ v8f wmma16(v16h a, v16h b, v8f c) {
  // D = A(16x32 f16) * B(32x16 f16) + C(16x16 f32)
  return __builtin_amdgcn_wmma_f32_16x16x32_f16(false, a, false, b, (short)0, c,
                                                false, false);
}

// A-fragment (16x32, 16-bit): lane l holds row M=l&15; lanes 0-15 carry
// K = {0..7,16..23}, lanes 16-31 carry K = {8..15,24..31}  (ISA 7.12.2).
// Per lane this is two contiguous 8xf16 runs -> two 128-bit loads.
static __device__ __forceinline__ v16h load_a16(const _Float16* p, int ld,
                                                int row0, int k0) {
  const int lane = threadIdx.x & 31;
  const _Float16* rp =
      p + (size_t)(row0 + (lane & 15)) * ld + k0 + ((lane >> 4) << 3);
  union { v16h v; v8h h[2]; } u;
  u.h[0] = *(const v8h*)rp;         // K offsets 0..7
  u.h[1] = *(const v8h*)(rp + 16);  // K offsets 16..23
  return u.v;
}

// B-fragment (32x16, 16-bit) from a PRE-TRANSPOSED weight ([n][k] layout):
// lane l holds col N=l&15; K = k0 + 8*(l>>4) + {0..7,16..23}  -> two
// contiguous 8xf16 runs per lane -> two 128-bit loads.
static __device__ __forceinline__ v16h load_bT(const _Float16* p, int ldk,
                                               int k0, int col0) {
  const int lane = threadIdx.x & 31;
  const _Float16* cp =
      p + (size_t)(col0 + (lane & 15)) * ldk + k0 + ((lane >> 4) << 3);
  union { v16h v; v8h h[2]; } u;
  u.h[0] = *(const v8h*)cp;
  u.h[1] = *(const v8h*)(cp + 16);
  return u.v;
}

// ---------------------------------------------------------------- conversions
__global__ void k_cvt_h(const float* __restrict__ in, _Float16* __restrict__ out,
                        int n) {
  int i = blockIdx.x * blockDim.x + threadIdx.x;
  if (i < n) out[i] = (_Float16)in[i];
}

// f32 [nmat][K][N] row-major  ->  f16 transposed [nmat][N][K]
__global__ void k_cvt_tr(const float* __restrict__ in, _Float16* __restrict__ out,
                         int K, int Ncols, int total) {
  int i = blockIdx.x * blockDim.x + threadIdx.x;
  if (i < total) {
    const int m = i / (K * Ncols);
    const int r = (i - m * K * Ncols) / Ncols;  // k
    const int c = i % Ncols;                    // n
    out[((size_t)m * Ncols + c) * K + r] = (_Float16)in[i];
  }
}

// ---------------------------------------------------------------- QKV GEMMs
// block = 4 waves, one 16-row tile of nodes; wave w owns 16-col tile w.
__global__ void k_qkv(const _Float16* __restrict__ embh,
                      const _Float16* __restrict__ qw, const float* __restrict__ qb,
                      const _Float16* __restrict__ kw, const float* __restrict__ kb,
                      const _Float16* __restrict__ vw, const float* __restrict__ vb,
                      _Float16* __restrict__ Qh, _Float16* __restrict__ Kh,
                      float* __restrict__ Vf) {
  const int row0 = blockIdx.x * 16;
  const int wave = threadIdx.x >> 5;
  const int lane = threadIdx.x & 31;
  const int n0   = wave * 16;
  const int col  = n0 + (lane & 15);
  const int rb   = row0 + 8 * (lane >> 4);

  v16h a0 = load_a16(embh, NDIM, row0, 0);
  v16h a1 = load_a16(embh, NDIM, row0, 32);

  {
    v8f c = {};
    c = wmma16(a0, load_bT(qw, NDIM, 0, n0), c);
    c = wmma16(a1, load_bT(qw, NDIM, 32, n0), c);
    const float bv = qb[col];
#pragma unroll
    for (int v = 0; v < 8; ++v)
      Qh[(size_t)(rb + v) * NDIM + col] = (_Float16)(c[v] + bv);
  }
  {
    v8f c = {};
    c = wmma16(a0, load_bT(kw, NDIM, 0, n0), c);
    c = wmma16(a1, load_bT(kw, NDIM, 32, n0), c);
    const float bv = kb[col];
#pragma unroll
    for (int v = 0; v < 8; ++v)
      Kh[(size_t)(rb + v) * NDIM + col] = (_Float16)(c[v] + bv);
  }
  {
    v8f c = {};
    c = wmma16(a0, load_bT(vw, NDIM, 0, n0), c);
    c = wmma16(a1, load_bT(vw, NDIM, 32, n0), c);
    const float bv = vb[col];
#pragma unroll
    for (int v = 0; v < 8; ++v)
      Vf[(size_t)(rb + v) * NDIM + col] = c[v] + bv;
  }
}

// ---------------------------------------------------------------- edge binning
__global__ void k_init(int* __restrict__ perm, int permN, int* __restrict__ bcnt,
                       int* __restrict__ rcnt, int nrows) {
  int i = blockIdx.x * blockDim.x + threadIdx.x;
  if (i < permN) perm[i] = -1;
  if (i < NBOND) bcnt[i] = 0;
  if (i < nrows) rcnt[i] = 0;
}

__global__ void k_hist_bond(const int* __restrict__ bond, int M, int TOT,
                            int* __restrict__ bcnt) {
  int e = blockIdx.x * blockDim.x + threadIdx.x;
  if (e < TOT) atomicAdd(&bcnt[bond[e < M ? e : e - M]], 1);
}

__global__ void k_scan_bond(const int* __restrict__ bcnt, int* __restrict__ boffp,
                            int* __restrict__ bcur) {
  if (blockIdx.x == 0 && threadIdx.x == 0) {
    int off = 0;
    for (int b = 0; b < NBOND; ++b) {
      boffp[b] = off;
      bcur[b]  = off;
      off += (bcnt[b] + 15) & ~15;   // pad each segment to 16-edge tiles
    }
    boffp[NBOND] = off;
  }
}

__global__ void k_scatter_bond(const int* __restrict__ bond, int M, int TOT,
                               int* __restrict__ bcur, int* __restrict__ perm) {
  int e = blockIdx.x * blockDim.x + threadIdx.x;
  if (e < TOT) {
    int bt  = bond[e < M ? e : e - M];
    int pos = atomicAdd(&bcur[bt], 1);
    perm[pos] = e;
  }
}

__global__ void k_hist_row(const int* __restrict__ src, const int* __restrict__ dst,
                           int M, int TOT, int* __restrict__ rcnt) {
  int e = blockIdx.x * blockDim.x + threadIdx.x;
  if (e < TOT) {
    int row = (e < M) ? src[e] : dst[e - M];
    atomicAdd(&rcnt[row], 1);
  }
}

__global__ void k_scan_row(const int* __restrict__ rcnt, int* __restrict__ roff,
                           int* __restrict__ rcur, int n) {
  if (blockIdx.x == 0 && threadIdx.x == 0) {
    int off = 0;
    for (int i = 0; i < n; ++i) {
      roff[i] = off;
      rcur[i] = off;
      off += rcnt[i];
    }
    roff[n] = off;
  }
}

__global__ void k_scatter_row(const int* __restrict__ src, const int* __restrict__ dst,
                              int M, int TOT, int* __restrict__ rcur,
                              int* __restrict__ redge) {
  int e = blockIdx.x * blockDim.x + threadIdx.x;
  if (e < TOT) {
    int row = (e < M) ? src[e] : dst[e - M];
    int pos = atomicAdd(&rcur[row], 1);
    redge[pos] = e;
  }
}

// ---------------------------------------------------------------- edge MLP
// One block = one 16-edge tile (uniform bond type); 4 waves = 4 heads.
// x(16x128) @ W0(128x64) -> relu -> @ W1(64x64) -> relu -> dot W2 -> leaky.
__global__ void k_edge_mlp(const _Float16* __restrict__ Qh,
                           const _Float16* __restrict__ Kh,
                           const _Float16* __restrict__ w0t, const float* __restrict__ b0,
                           const _Float16* __restrict__ w1t, const float* __restrict__ b1,
                           const float* __restrict__ W2, const float* __restrict__ b2,
                           const int* __restrict__ src, const int* __restrict__ dst,
                           const int* __restrict__ perm, const int* __restrict__ boffp,
                           float* __restrict__ scores, int M) {
  __shared__ __align__(16) _Float16 xs[16][2 * NDIM];     // edge features (A op)
  __shared__ __align__(16) _Float16 h0s[NHEAD][16][HIDD]; // relu(h0) -> next A
  __shared__ __align__(16) float    h1s[NHEAD][16][HIDD]; // relu(h1) f32
  __shared__ int eS[16], aS[16], bS[16];

  const int base = blockIdx.x * 16;
  if (base >= boffp[NBOND]) return;              // uniform exit
  int bt = 0;
#pragma unroll
  for (int b = 1; b < NBOND; ++b)
    if (base >= boffp[b]) bt = b;

  const int tid = threadIdx.x;
  const _Float16* w0p_all = w0t + (size_t)(bt * NHEAD) * (2 * NDIM) * HIDD;
  const _Float16* w1p_all = w1t + (size_t)(bt * NHEAD) * HIDD * HIDD;
  __builtin_prefetch(w0p_all, 0, 3);
  __builtin_prefetch(w1p_all, 0, 3);

  if (tid < 16) {
    int e = perm[base + tid];
    int a = 0, d = 0;
    if (e >= 0) {
      if (e < M) { a = src[e];     d = dst[e]; }
      else       { a = dst[e - M]; d = src[e - M]; }
    }
    eS[tid] = e; aS[tid] = a; bS[tid] = d;
  }
  __syncthreads();

  {  // stage x = [Q[sa] | K[db]] : 16 rows x 128 cols = 256 chunks of 8xf16
#pragma unroll
    for (int it = 0; it < 2; ++it) {
      const int ci  = tid + it * 128;
      const int r   = ci >> 4;
      const int seg = ci & 15;
      const _Float16* sp = (seg < 8)
          ? (Qh + (size_t)aS[r] * NDIM + seg * 8)
          : (Kh + (size_t)bS[r] * NDIM + (seg - 8) * 8);
#if USE_ASYNC_LDS
      async_cp16(&xs[r][seg * 8], sp);            // ASYNCcnt-tracked DMA to LDS
#else
      *(v8h*)&xs[r][seg * 8] = *(const v8h*)sp;
#endif
    }
#if USE_ASYNC_LDS
    __builtin_amdgcn_s_wait_asynccnt(0);
#endif
  }
  __syncthreads();

  const int h     = tid >> 5;          // wave == head
  const int lane  = tid & 31;
  const int col15 = lane & 15;
  const int rhalf = 8 * (lane >> 4);

  v16h ax[4];
#pragma unroll
  for (int kc = 0; kc < 4; ++kc) ax[kc] = load_a16(&xs[0][0], 2 * NDIM, 0, kc * 32);

  const _Float16* w0h_p = w0t + (size_t)(bt * NHEAD + h) * (2 * NDIM) * HIDD;
  const float*    b0p   = b0 + (bt * NHEAD + h) * HIDD;
#pragma unroll
  for (int nt = 0; nt < 4; ++nt) {
    v8f c = {};
#pragma unroll
    for (int kc = 0; kc < 4; ++kc)
      c = wmma16(ax[kc], load_bT(w0h_p, 2 * NDIM, kc * 32, nt * 16), c);
    const float bv = b0p[nt * 16 + col15];
#pragma unroll
    for (int v = 0; v < 8; ++v)
      h0s[h][v + rhalf][nt * 16 + col15] = (_Float16)fmaxf(c[v] + bv, 0.0f);
  }
  // h0s producer/consumer is same-wave; per-wave LDS ordering + compiler waits.
  v16h ah[2];
#pragma unroll
  for (int kc = 0; kc < 2; ++kc) ah[kc] = load_a16(&h0s[h][0][0], HIDD, 0, kc * 32);

  const _Float16* w1h_p = w1t + (size_t)(bt * NHEAD + h) * HIDD * HIDD;
  const float*    b1p   = b1 + (bt * NHEAD + h) * HIDD;
#pragma unroll
  for (int nt = 0; nt < 4; ++nt) {
    v8f c = {};
#pragma unroll
    for (int kc = 0; kc < 2; ++kc)
      c = wmma16(ah[kc], load_bT(w1h_p, HIDD, kc * 32, nt * 16), c);
    const float bv = b1p[nt * 16 + col15];
#pragma unroll
    for (int v = 0; v < 8; ++v)
      h1s[h][v + rhalf][nt * 16 + col15] = fmaxf(c[v] + bv, 0.0f);
  }
  __syncthreads();

  if (tid < 64) {                       // 4 heads x 16 edges
    const int hh = tid >> 4, j = tid & 15;
    const int e = eS[j];
    if (e >= 0) {
      const float* w2p = W2 + (bt * NHEAD + hh) * HIDD;
      float s = b2[bt * NHEAD + hh];
#pragma unroll 8
      for (int k = 0; k < HIDD; ++k) s += h1s[hh][j][k] * w2p[k];
      s = (s > 0.0f) ? s : NEG_SLOPE * s;
      scores[(size_t)e * NHEAD + hh] = s;
    }
  }
}

// ---------------------------------------------------------------- sparse softmax + A@V
__global__ void k_softmax_agg(const float* __restrict__ scores,
                              const int* __restrict__ roff,
                              const int* __restrict__ redge,
                              const int* __restrict__ src, const int* __restrict__ dst,
                              const float* __restrict__ Vf,
                              _Float16* __restrict__ Oh, int M) {
  const int i    = blockIdx.x;
  const int h    = threadIdx.x >> 5;
  const int lane = threadIdx.x & 31;
  const int e0 = roff[i], e1 = roff[i + 1];

  float m = -1e30f;
  for (int p = e0 + lane; p < e1; p += 32)
    m = fmaxf(m, scores[(size_t)redge[p] * NHEAD + h]);
#pragma unroll
  for (int o = 16; o; o >>= 1) m = fmaxf(m, __shfl_xor(m, o));

  float z = 0.0f;
  for (int p = e0 + lane; p < e1; p += 32)
    z += __expf(scores[(size_t)redge[p] * NHEAD + h] - m);
#pragma unroll
  for (int o = 16; o; o >>= 1) z += __shfl_xor(z, o);
  const float inv = (z > 0.0f) ? 1.0f / z : 0.0f;

  float o0 = 0.0f, o1 = 0.0f;
  for (int p = e0; p < e1; ++p) {
    const int e = redge[p];
    const float wgt = __expf(scores[(size_t)e * NHEAD + h] - m) * inv;
    const int j = (e < M) ? dst[e] : src[e - M];
    o0 += wgt * Vf[(size_t)j * NDIM + lane];
    o1 += wgt * Vf[(size_t)j * NDIM + 32 + lane];
  }
  Oh[(size_t)i * (NHEAD * NDIM) + h * NDIM + lane]      = (_Float16)o0;
  Oh[(size_t)i * (NHEAD * NDIM) + h * NDIM + 32 + lane] = (_Float16)o1;
}

// ---------------------------------------------------------------- final projection
__global__ void k_proj(const _Float16* __restrict__ Oh,
                       const _Float16* __restrict__ pwt,
                       const float* __restrict__ pb, float* __restrict__ out) {
  const int row0 = blockIdx.x * 16;
  const int wave = threadIdx.x >> 5;
  const int lane = threadIdx.x & 31;
  const int n0   = wave * 16;
  v8f c = {};
#pragma unroll
  for (int kc = 0; kc < 8; ++kc)
    c = wmma16(load_a16(Oh, NHEAD * NDIM, row0, kc * 32),
               load_bT(pwt, NHEAD * NDIM, kc * 32, n0), c);
  const int col = n0 + (lane & 15);
  const int rb  = row0 + 8 * (lane >> 4);
  const float bv = pb[col];
#pragma unroll
  for (int v = 0; v < 8; ++v) out[(size_t)(rb + v) * NDIM + col] = c[v] + bv;
}

// ---------------------------------------------------------------- launcher
extern "C" void kernel_launch(void* const* d_in, const int* in_sizes, int n_in,
                              void* d_out, int out_size, void* d_ws, size_t ws_size,
                              hipStream_t stream) {
  (void)n_in; (void)out_size; (void)ws_size;
  const float* emb = (const float*)d_in[0];
  const float* Qw  = (const float*)d_in[1];
  const float* Qb  = (const float*)d_in[2];
  const float* Kw  = (const float*)d_in[3];
  const float* Kb  = (const float*)d_in[4];
  const float* Vw  = (const float*)d_in[5];
  const float* Vb  = (const float*)d_in[6];
  const float* W0  = (const float*)d_in[7];
  const float* b0  = (const float*)d_in[8];
  const float* W1  = (const float*)d_in[9];
  const float* b1  = (const float*)d_in[10];
  const float* W2  = (const float*)d_in[11];
  const float* b2  = (const float*)d_in[12];
  const float* Pw  = (const float*)d_in[13];
  const float* Pb  = (const float*)d_in[14];
  const int* src   = (const int*)d_in[15];
  const int* dst   = (const int*)d_in[16];
  const int* bond  = (const int*)d_in[17];

  const int N     = in_sizes[0] / NDIM;   // 4096
  const int M     = in_sizes[15];         // E + N directed-source edges
  const int TOT   = 2 * M;                // symmetric directed edges
  const int PERMN = TOT + 16 * NBOND;     // per-bond tile padding

  uint8_t* wsb = (uint8_t*)d_ws;
  size_t off = 0;
  auto alloc = [&](size_t bytes) -> void* {
    off = (off + 255) & ~(size_t)255;
    void* p = (void*)(wsb + off);
    off += bytes;
    return p;
  };

  _Float16* embh = (_Float16*)alloc((size_t)N * NDIM * 2);
  _Float16* qwt  = (_Float16*)alloc((size_t)NDIM * NDIM * 2);     // [n][k]
  _Float16* kwt  = (_Float16*)alloc((size_t)NDIM * NDIM * 2);
  _Float16* vwt  = (_Float16*)alloc((size_t)NDIM * NDIM * 2);
  _Float16* w0t  = (_Float16*)alloc((size_t)NBOND * NHEAD * 2 * NDIM * HIDD * 2);
  _Float16* w1t  = (_Float16*)alloc((size_t)NBOND * NHEAD * HIDD * HIDD * 2);
  _Float16* pwt  = (_Float16*)alloc((size_t)NHEAD * NDIM * NDIM * 2);
  _Float16* Qh   = (_Float16*)alloc((size_t)N * NDIM * 2);
  _Float16* Kh   = (_Float16*)alloc((size_t)N * NDIM * 2);
  float*    Vf   = (float*)alloc((size_t)N * NDIM * 4);
  float*    scr  = (float*)alloc((size_t)TOT * NHEAD * 4);
  int*      perm = (int*)alloc((size_t)PERMN * 4);
  int*      bcnt = (int*)alloc(NBOND * 4);
  int*      boffp= (int*)alloc((NBOND + 1) * 4);
  int*      bcur = (int*)alloc(NBOND * 4);
  int*      rcnt = (int*)alloc((size_t)N * 4);
  int*      roff = (int*)alloc((size_t)(N + 1) * 4);
  int*      rcur = (int*)alloc((size_t)N * 4);
  int*      redge= (int*)alloc((size_t)TOT * 4);
  _Float16* Oh   = (_Float16*)alloc((size_t)N * NHEAD * NDIM * 2);

  const int CT = 256;
  auto g = [&](int n) { return dim3((n + CT - 1) / CT); };

  // fp32 -> fp16 conversions; weights transposed to [n][k] for contiguous
  // B-fragment loads.
  k_cvt_h<<<g(N * NDIM), CT, 0, stream>>>(emb, embh, N * NDIM);
  k_cvt_tr<<<g(NDIM * NDIM), CT, 0, stream>>>(Qw, qwt, NDIM, NDIM, NDIM * NDIM);
  k_cvt_tr<<<g(NDIM * NDIM), CT, 0, stream>>>(Kw, kwt, NDIM, NDIM, NDIM * NDIM);
  k_cvt_tr<<<g(NDIM * NDIM), CT, 0, stream>>>(Vw, vwt, NDIM, NDIM, NDIM * NDIM);
  {
    const int t0 = NBOND * NHEAD * 2 * NDIM * HIDD;
    k_cvt_tr<<<g(t0), CT, 0, stream>>>(W0, w0t, 2 * NDIM, HIDD, t0);
    const int t1 = NBOND * NHEAD * HIDD * HIDD;
    k_cvt_tr<<<g(t1), CT, 0, stream>>>(W1, w1t, HIDD, HIDD, t1);
    const int tp = NHEAD * NDIM * NDIM;
    k_cvt_tr<<<g(tp), CT, 0, stream>>>(Pw, pwt, NHEAD * NDIM, NDIM, tp);
  }

  // QKV (WMMA)
  k_qkv<<<dim3(N / 16), 128, 0, stream>>>(embh, qwt, Qb, kwt, Kb, vwt, Vb, Qh, Kh, Vf);

  // bond-type counting sort (tile-padded) + row CSR
  int initN = PERMN > N ? PERMN : N;
  k_init<<<g(initN), CT, 0, stream>>>(perm, PERMN, bcnt, rcnt, N);
  k_hist_bond<<<g(TOT), CT, 0, stream>>>(bond, M, TOT, bcnt);
  k_scan_bond<<<dim3(1), 1, 0, stream>>>(bcnt, boffp, bcur);
  k_scatter_bond<<<g(TOT), CT, 0, stream>>>(bond, M, TOT, bcur, perm);
  k_hist_row<<<g(TOT), CT, 0, stream>>>(src, dst, M, TOT, rcnt);
  k_scan_row<<<dim3(1), 1, 0, stream>>>(rcnt, roff, rcur, N);
  k_scatter_row<<<g(TOT), CT, 0, stream>>>(src, dst, M, TOT, rcur, redge);

  // edge MLP (WMMA, dominant compute)
  const int ntiles = (TOT + 16 * NBOND + 15) / 16;
  k_edge_mlp<<<dim3(ntiles), 128, 0, stream>>>(Qh, Kh, w0t, b0, w1t, b1, W2, b2,
                                               src, dst, perm, boffp, scr, M);

  // sparse softmax + aggregation
  k_softmax_agg<<<dim3(N), 128, 0, stream>>>(scr, roff, redge, src, dst, Vf, Oh, M);

  // output projection (WMMA)
  k_proj<<<dim3(N / 16), 128, 0, stream>>>(Oh, pwt, Pb, (float*)d_out);
}
